// LMLTVIT_33174327394726
// MI455X (gfx1250) — matrix operand order repaired and checked by reading, not compiled, not run on hardware
//
#include <hip/hip_runtime.h>
#include <hip/hip_bf16.h>

// ---------------------------------------------------------------------------
// Fully fused LMLT-ViT window-attention block for gfx1250 (MI455X).
// One workgroup (128 threads = 4 wave32) per 8x8 window; 16384 windows.
//   conv(OmniShift, combined 5x5) -> QKV -> softmax(QK^T) -> PV + LePE -> proj
// All matmuls via v_wmma_f32_16x16x32_f16; fragments are aligned ds_load_b128
// (B operands K-major in LDS; OOB K groups via branchless zero-pad cndmask).
// Conv uses channel-planar halo + register-strip reuse (v4h LDS loads,
// v_fma_mix f16xf32 FMAs). WMMA tile loops fully unrolled.
// ---------------------------------------------------------------------------

typedef _Float16 v16h __attribute__((ext_vector_type(16)));
typedef _Float16 v8h  __attribute__((ext_vector_type(8)));
typedef _Float16 v4h  __attribute__((ext_vector_type(4)));
typedef float    v8f  __attribute__((ext_vector_type(8)));

union FFrag { v8f v; float f[8]; };

#define B_DIM   4
#define HW      512
#define C_DIM   48
#define WS      8
#define NTOK    64
#define NWIN    16384           // 4 * (512/8)^2
#define HALO    12
#define QKV_N   144

__device__ __forceinline__ v16h cat16(v8h lo, v8h hi) {
  return __builtin_shufflevector(lo, hi, 0, 1, 2, 3, 4, 5, 6, 7,
                                 8, 9, 10, 11, 12, 13, 14, 15);
}

// A-matrix 16x32 f16 fragment from row-major [M][K] tile (K contiguous).
template <int LD, int K0, int KMAX>
__device__ __forceinline__ v16h load_a(const _Float16* sm, int row0, int lane,
                                       const _Float16* zb) {
  const int kb = K0 + ((lane >> 4) << 4);
  const _Float16* rp = sm + (row0 + (lane & 15)) * LD;
  const _Float16* p0 = (kb + 16 <= KMAX) ? (rp + kb) : zb;
  const _Float16* p1 = (kb + 16 <= KMAX) ? (rp + kb + 8) : zb;
  return cat16(*(const v8h*)p0, *(const v8h*)p1);
}

// B-matrix 32x16 f16 fragment from K-major [N][K] tile (K contiguous).
template <int LD, int K0, int KMAX>
__device__ __forceinline__ v16h load_bk(const _Float16* sm, int col0, int lane,
                                        const _Float16* zb) {
  const int kb = K0 + ((lane >> 4) << 4);
  const _Float16* rp = sm + (col0 + (lane & 15)) * LD + kb;
  if (K0 + 32 > KMAX) rp = (kb + 16 <= KMAX) ? rp : zb;
  return cat16(*(const v8h*)rp, *(const v8h*)(rp + 8));
}

__device__ __forceinline__ v8f wmma_f16(v16h a, v16h b, v8f c) {
  return __builtin_amdgcn_wmma_f32_16x16x32_f16(false, a, false, b,
                                                (short)0, c, false, false);
}

// ---- shared memory: phase-based unions, ~58 KB ----------------------------
struct SMem {
  union {                                      // 18624 B
    struct {                                   //   phases A-B
      alignas(16) _Float16 xh[C_DIM * HALO * HALO]; // halo, channel-planar
      float wcomb[C_DIM * 25];                 //   combined 5x5 filter
    } pre;
    float ss[NTOK * NTOK];                     //   D-E: scores; F-G: O (f32)
  } u1;
  union {                                      // 6144 B
    alignas(16) _Float16 xs[NTOK * C_DIM];     //   B-C: conv out, row-major
    alignas(16) _Float16 osh[NTOK * C_DIM];    //   G-H: O + lepe, row-major
  } u2;
  union {                                      // 13824 B
    alignas(16) _Float16 wqkvT[QKV_N * C_DIM]; //   A-C: qkv weights, K-major
    alignas(16) _Float16 ps[NTOK * NTOK];      //   E-F: softmax probs
    alignas(16) _Float16 wpT[C_DIM * C_DIM];   //   G-H: proj weights, K-major
  } u3;
  alignas(16) _Float16 qs[NTOK * C_DIM];       // [tok][ch], scale folded
  alignas(16) _Float16 ks[NTOK * C_DIM];       // [tok][ch] (== K^T K-major)
  alignas(16) _Float16 vsT[C_DIM * NTOK];      // [ch][tok] (B for P@V; lepe)
  alignas(16) _Float16 zpad[16];               // zero pad for OOB K
  float bqkv[QKV_N];
  float bp[C_DIM];
  float wv[C_DIM * 9];
  float bv[C_DIM];
};

__global__ __launch_bounds__(128)
void lmlt_vit_fused_kernel(const float* __restrict__ x,
                           const float* __restrict__ w1,
                           const float* __restrict__ w3,
                           const float* __restrict__ w5,
                           const float* __restrict__ alpha,
                           const float* __restrict__ qkv_w,
                           const float* __restrict__ qkv_b,
                           const float* __restrict__ get_v_w,
                           const float* __restrict__ get_v_b,
                           const float* __restrict__ proj_w,
                           const float* __restrict__ proj_b,
                           float* __restrict__ out) {
  __shared__ SMem sm;

  const int tid  = threadIdx.x;
  const int lane = tid & 31;
  const int wave = tid >> 5;

  const int win = blockIdx.x;
  const int b   = win >> 12;
  const int rem = win & 4095;
  const int wy  = rem >> 6;
  const int wx  = rem & 63;
  const int h0  = wy * WS - 2;
  const int w0  = wx * WS - 2;

  const float a0 = alpha[0], a1 = alpha[1], a2 = alpha[2], a3 = alpha[3];
  const float scale = 0.14433756729740643f;  // 48^-0.5
  const _Float16* zb = sm.zpad;

  // ---- Phase A: halo tile (channel-planar), weights, combined filter ------
  for (int i = tid; i < HALO * HALO * C_DIM; i += 128) {
    int c  = i % C_DIM;                      // coalesced global read
    int p  = i / C_DIM;
    int xx = p % HALO;
    int yy = p / HALO;
    int hh = h0 + yy, ww = w0 + xx;
    float v = 0.0f;
    if ((unsigned)hh < HW && (unsigned)ww < HW)
      v = x[((size_t)(b * HW + hh) * HW + ww) * C_DIM + c];
    sm.u1.pre.xh[c * (HALO * HALO) + p] = (_Float16)v;   // planar LDS write
  }
  for (int i = tid; i < C_DIM * 25; i += 128) {
    int c = i / 25, j = i % 25;
    int dy = j / 5, dx = j % 5;
    float v = a3 * w5[i];
    if (dy >= 1 && dy <= 3 && dx >= 1 && dx <= 3)
      v += a2 * w3[c * 9 + (dy - 1) * 3 + (dx - 1)];
    if (j == 12) v += a0 + a1 * w1[c];
    sm.u1.pre.wcomb[i] = v;
  }
  // qkv_w [k][n] -> wqkvT [n][k]
  for (int i = tid; i < C_DIM * QKV_N; i += 128)
    sm.u3.wqkvT[(i % QKV_N) * C_DIM + (i / QKV_N)] = (_Float16)qkv_w[i];
  for (int i = tid; i < QKV_N; i += 128) sm.bqkv[i] = qkv_b[i];
  for (int i = tid; i < C_DIM * 9; i += 128) sm.wv[i] = get_v_w[i];
  if (tid < C_DIM) { sm.bp[tid] = proj_b[tid]; sm.bv[tid] = get_v_b[tid]; }
  if (tid < 16) sm.zpad[tid] = (_Float16)0.0f;
  __syncthreads();

  // ---- Phase B: combined 5x5 depthwise conv, register-strip form ----------
  // 384 strips = 48 ch x 8 output rows; each strip: 8 outputs, 5x12 halo rows
  // kept in registers (3x v4h loads per row), 25 taps reused across outputs.
#pragma unroll
  for (int si = 0; si < 3; ++si) {
    int s = tid + si * 128;
    int c = s % C_DIM;
    int y = s / C_DIM;
    const float* wk = &sm.u1.pre.wcomb[c * 25];
    float acc[8];
#pragma unroll
    for (int xw = 0; xw < 8; ++xw) acc[xw] = 0.0f;
#pragma unroll
    for (int dy = 0; dy < 5; ++dy) {
      const _Float16* rp = &sm.u1.pre.xh[c * (HALO * HALO) + (y + dy) * HALO];
      union { v4h v[3]; _Float16 h[12]; } row;
      row.v[0] = *(const v4h*)(rp);
      row.v[1] = *(const v4h*)(rp + 4);
      row.v[2] = *(const v4h*)(rp + 8);
#pragma unroll
      for (int dx = 0; dx < 5; ++dx) {
        float w = wk[dy * 5 + dx];
#pragma unroll
        for (int xw = 0; xw < 8; ++xw)
          acc[xw] += w * (float)row.h[xw + dx];      // v_fma_mix_f32
      }
    }
#pragma unroll
    for (int xw = 0; xw < 8; ++xw)
      sm.u2.xs[(y * 8 + xw) * C_DIM + c] = (_Float16)acc[xw];
  }
  __syncthreads();

  // ---- Phase C: QKV = Xs @ Wqkv + b  (4x9 tiles, 9 per wave, unrolled) ----
#pragma unroll
  for (int it = 0; it < 9; ++it) {
    int t  = wave + it * 4;
    int mi = t / 9, nj = t % 9;
    FFrag acc; acc.v = (v8f){};
    acc.v = wmma_f16(load_a<C_DIM, 0, C_DIM>(sm.u2.xs, mi * 16, lane, zb),
                     load_bk<C_DIM, 0, C_DIM>(sm.u3.wqkvT, nj * 16, lane, zb),
                     acc.v);
    acc.v = wmma_f16(load_a<C_DIM, 32, C_DIM>(sm.u2.xs, mi * 16, lane, zb),
                     load_bk<C_DIM, 32, C_DIM>(sm.u3.wqkvT, nj * 16, lane, zb),
                     acc.v);
    int n    = nj * 16 + (lane & 15);
    int rowb = mi * 16 + ((lane >> 4) << 3);
    int seg  = __builtin_amdgcn_readfirstlane(nj / 3);  // 0=Q 1=K 2=V uniform
    int cloc = (nj % 3) * 16 + (lane & 15);
    float bias = sm.bqkv[n];
    if (seg == 0) {
#pragma unroll
      for (int r = 0; r < 8; ++r)
        sm.qs[(rowb + r) * C_DIM + cloc] = (_Float16)((acc.f[r] + bias) * scale);
    } else if (seg == 1) {
#pragma unroll
      for (int r = 0; r < 8; ++r)
        sm.ks[(rowb + r) * C_DIM + cloc] = (_Float16)(acc.f[r] + bias);
    } else {
#pragma unroll
      for (int r = 0; r < 8; ++r)
        sm.vsT[cloc * NTOK + (rowb + r)] = (_Float16)(acc.f[r] + bias);
    }
  }
  __syncthreads();

  // ---- Phase D: S = (scale*Q) @ K^T  (4x4 tiles, 4 per wave, unrolled) ----
#pragma unroll
  for (int it = 0; it < 4; ++it) {
    int t  = wave + it * 4;
    int mi = t >> 2, nj = t & 3;
    FFrag acc; acc.v = (v8f){};
    acc.v = wmma_f16(load_a<C_DIM, 0, C_DIM>(sm.qs, mi * 16, lane, zb),
                     load_bk<C_DIM, 0, C_DIM>(sm.ks, nj * 16, lane, zb),
                     acc.v);
    acc.v = wmma_f16(load_a<C_DIM, 32, C_DIM>(sm.qs, mi * 16, lane, zb),
                     load_bk<C_DIM, 32, C_DIM>(sm.ks, nj * 16, lane, zb),
                     acc.v);
    int n    = nj * 16 + (lane & 15);
    int rowb = mi * 16 + ((lane >> 4) << 3);
#pragma unroll
    for (int r = 0; r < 8; ++r)
      sm.u1.ss[(rowb + r) * NTOK + n] = acc.f[r];
  }
  __syncthreads();

  // ---- Phase E: row softmax -> Ps (f16) -----------------------------------
  if (tid < NTOK) {
    const float* srow = &sm.u1.ss[tid * NTOK];
    float m = srow[0];
#pragma unroll 8
    for (int j = 1; j < NTOK; ++j) m = fmaxf(m, srow[j]);
    float s = 0.0f;
#pragma unroll 8
    for (int j = 0; j < NTOK; ++j) s += __expf(srow[j] - m);
    float inv = 1.0f / s;
#pragma unroll 8
    for (int j = 0; j < NTOK; ++j)
      sm.u3.ps[tid * NTOK + j] = (_Float16)(__expf(srow[j] - m) * inv);
  }
  __syncthreads();

  // ---- Phase F: O = P @ V  (4x3 tiles, 3 per wave, unrolled) --------------
#pragma unroll
  for (int it = 0; it < 3; ++it) {
    int t  = wave + it * 4;
    int mi = t / 3, nj = t % 3;
    FFrag acc; acc.v = (v8f){};
    acc.v = wmma_f16(load_a<NTOK, 0, NTOK>(sm.u3.ps, mi * 16, lane, zb),
                     load_bk<NTOK, 0, NTOK>(sm.vsT, nj * 16, lane, zb),
                     acc.v);
    acc.v = wmma_f16(load_a<NTOK, 32, NTOK>(sm.u3.ps, mi * 16, lane, zb),
                     load_bk<NTOK, 32, NTOK>(sm.vsT, nj * 16, lane, zb),
                     acc.v);
    int n    = nj * 16 + (lane & 15);
    int rowb = mi * 16 + ((lane >> 4) << 3);
#pragma unroll
    for (int r = 0; r < 8; ++r)
      sm.u1.ss[(rowb + r) * NTOK + n] = acc.f[r];
  }
  __syncthreads();

  // ---- Phase G: lepe = dwconv3x3(V)+bv ; Osh = O + lepe ; stage WpT -------
  for (int i = tid; i < C_DIM * C_DIM; i += 128)
    sm.u3.wpT[(i % C_DIM) * C_DIM + (i / C_DIM)] = (_Float16)proj_w[i];
  for (int i = tid; i < NTOK * C_DIM; i += 128) {
    int c = i % C_DIM;
    int n = i / C_DIM;
    int y = n >> 3, xw = n & 7;
    float acc = sm.bv[c];
#pragma unroll
    for (int dy = 0; dy < 3; ++dy) {
      int yy = y + dy - 1;
      if ((unsigned)yy >= WS) continue;
#pragma unroll
      for (int dx = 0; dx < 3; ++dx) {
        int xx = xw + dx - 1;
        if ((unsigned)xx >= WS) continue;
        acc += sm.wv[c * 9 + dy * 3 + dx] *
               (float)sm.vsT[c * NTOK + (yy * WS + xx)];
      }
    }
    sm.u2.osh[n * C_DIM + c] = (_Float16)(sm.u1.ss[n * NTOK + c] + acc);
  }
  __syncthreads();

  // ---- Phase H: out = Osh @ Wp + bp, window-reverse scatter ---------------
#pragma unroll
  for (int it = 0; it < 3; ++it) {
    int t  = wave + it * 4;
    int mi = t / 3, nj = t % 3;
    FFrag acc; acc.v = (v8f){};
    acc.v = wmma_f16(load_a<C_DIM, 0, C_DIM>(sm.u2.osh, mi * 16, lane, zb),
                     load_bk<C_DIM, 0, C_DIM>(sm.u3.wpT, nj * 16, lane, zb),
                     acc.v);
    acc.v = wmma_f16(load_a<C_DIM, 32, C_DIM>(sm.u2.osh, mi * 16, lane, zb),
                     load_bk<C_DIM, 32, C_DIM>(sm.u3.wpT, nj * 16, lane, zb),
                     acc.v);
    int c    = nj * 16 + (lane & 15);
    int rowb = mi * 16 + ((lane >> 4) << 3);
    float bias = sm.bp[c];
#pragma unroll
    for (int r = 0; r < 8; ++r) {
      int n = rowb + r;
      int y = n >> 3, xw = n & 7;
      int hh = wy * WS + y, ww = wx * WS + xw;
      out[((size_t)(b * HW + hh) * HW + ww) * C_DIM + c] = acc.f[r] + bias;
    }
  }
}

extern "C" void kernel_launch(void* const* d_in, const int* in_sizes, int n_in,
                              void* d_out, int out_size, void* d_ws, size_t ws_size,
                              hipStream_t stream) {
  (void)in_sizes; (void)n_in; (void)out_size; (void)d_ws; (void)ws_size;
  const float* x       = (const float*)d_in[0];
  // d_in[1] = H, d_in[2] = W (int scalars; dims fixed: 4 x 512 x 512 x 48)
  const float* w1      = (const float*)d_in[3];
  const float* w3      = (const float*)d_in[4];
  const float* w5      = (const float*)d_in[5];
  const float* alpha   = (const float*)d_in[6];
  const float* qkv_w   = (const float*)d_in[7];
  const float* qkv_b   = (const float*)d_in[8];
  const float* get_v_w = (const float*)d_in[9];
  const float* get_v_b = (const float*)d_in[10];
  const float* proj_w  = (const float*)d_in[11];
  const float* proj_b  = (const float*)d_in[12];
  float* out = (float*)d_out;

  lmlt_vit_fused_kernel<<<NWIN, 128, 0, stream>>>(
      x, w1, w3, w5, alpha, qkv_w, qkv_b, get_v_w, get_v_b, proj_w, proj_b, out);
}
